// PillarAttentionSampling_83837761618200
// MI455X (gfx1250) — matrix operand order repaired and checked by reading, not compiled, not run on hardware
//
#include <hip/hip_runtime.h>
#include <hip/hip_bf16.h>
#include <math.h>

// MI455X / gfx1250, wave32. Strength-reduced pillar attention sampling.
//
// Shapes: B=1,H=8,W=4 -> 32 cells; P=HD=128; C=4; HEADS=4; HID=512; K_SEL=64.
// attn_dot[p] = (1/sqrt(128)) * sum_{m,n} G[p,m,n]*T[p,m,n]
//             + NEG * sum_n S[p,n]*Wneg[n] + w_b
// with rank-5 augmented x~ (bias folded):
//   G[p,m,n] = sum_{c,c'} x~[p,c] x~[p,c'] GK[c*4+m][c'*4+n]   (GK: 20x20 Gram, K=128, cell-invariant)
//   T[p,m,n] = sum_c x~[p,c] QT[c*4+m][n]                      (QT: 20x4 masked matmul, K=128, per cell)
//   S[p,n]   = sum_c x~[p,c] VS[c*4+n]
// GK and QT are computed with V_WMMA_F32_16X16X32_F16 (4 k-steps of 32).

typedef __attribute__((ext_vector_type(16))) _Float16 v16h;
typedef __attribute__((ext_vector_type(8)))  float    v8f;

#define P_     128
#define C_     4
#define HID_   512
#define HD_    128
#define KSEL_  64
#define NCELL_ 32
#define NEG_   (-1e9f)
#define THRESH_ 0.7f

// ---- WMMA fragment gathers from LDS (ISA 7.12.2 16-bit layouts, wave32) ----

// A is 16x32 f16, LDS row-major [rows][lda]. Lane<16: M=lane, VGPR r(0..3)->K=2r,2r+1,
// VGPR r(4..7)->K=16+2(r-4); lanes 16..31 add +8 to K base.
__device__ inline v16h ldsA_frag(const _Float16* A, int lda, int row0, int k0, int lane) {
  const int h = (lane >> 4) & 1, m = lane & 15;
  v16h a;
#pragma unroll
  for (int r = 0; r < 8; ++r) {
    const int kb = k0 + 2 * (r & 3) + 16 * (r >> 2) + 8 * h;
    const _Float16* row = A + (row0 + m) * lda;
    a[2 * r]     = row[kb];
    a[2 * r + 1] = row[kb + 1];
  }
  return a;
}

// Math B is [K=32 x N=16]; stored in LDS transposed as Bm[N][K] (row-major, ld=ldb).
// Lane<16: N=lane, VGPR r(0..3)->K=2r,2r+1, VGPR r(4..7)->K=16+2(r-4); lanes 16..31 add +8 to K base.
__device__ inline v16h ldsBT_frag(const _Float16* Bm, int ldb, int n0, int k0, int lane) {
  const int h = (lane >> 4) & 1, n = lane & 15;
  v16h b;
#pragma unroll
  for (int r = 0; r < 8; ++r) {
    const int k = k0 + 2 * (r & 3) + 16 * (r >> 2) + 8 * h;
    const _Float16* row = Bm + (n0 + n) * ldb;
    b[2 * r]     = row[k];
    b[2 * r + 1] = row[k + 1];
  }
  return b;
}

// ---------------- Kernel 1: cell-invariant precompute ----------------
// GK[20][20] = k~w . v~w^T (contraction over j=128 within each head block),
// VS[20]     = row sums of v~w. Stored padded to 32 in workspace.
__global__ __launch_bounds__(64) void pillar_precompute(
    const float* __restrict__ kw, const float* __restrict__ kb,
    const float* __restrict__ vw, const float* __restrict__ vb,
    float* __restrict__ gk_ws, float* __restrict__ vs_ws) {
  __shared__ _Float16 Ak[32][128];  // rows cm = c*4+m (c<5 incl. bias row), cols j
  __shared__ _Float16 Bv[32][128];
  const int t = threadIdx.x;

  {  // fill: 64 threads, each covers half a row
    const int row = t & 31, jb = (t >> 5) * 64;
    const int c = row >> 2, m = row & 3;
    for (int j = jb; j < jb + 64; ++j) {
      float av = 0.f, bv = 0.f;
      if (row < 20) {
        av = (c < 4) ? kw[c * HID_ + m * HD_ + j] : kb[m * HD_ + j];
        bv = (c < 4) ? vw[c * HID_ + m * HD_ + j] : vb[m * HD_ + j];
      }
      Ak[row][j] = (_Float16)av;
      Bv[row][j] = (_Float16)bv;
    }
  }
  __syncthreads();

  if (t < 32) {  // wave 0: 2x2 output tiles, K=128 in 4 steps -> 16 WMMAs
    const int lane = t;
#pragma unroll
    for (int mt = 0; mt < 2; ++mt) {
#pragma unroll
      for (int nt = 0; nt < 2; ++nt) {
        v8f acc = {};
#pragma unroll
        for (int kc = 0; kc < 4; ++kc) {
          v16h a = ldsA_frag(&Ak[0][0], 128, mt * 16, kc * 32, lane);
          v16h b = ldsBT_frag(&Bv[0][0], 128, nt * 16, kc * 32, lane);
          acc = __builtin_amdgcn_wmma_f32_16x16x32_f16(false, a, false, b,
                                                       (short)0, acc, false, false);
        }
        const int h = lane >> 4, n = lane & 15;
#pragma unroll
        for (int r = 0; r < 8; ++r)
          gk_ws[(mt * 16 + r + 8 * h) * 32 + nt * 16 + n] = acc[r];
      }
    }
  } else {  // wave 1: VS row sums (rows >= 20 are zero-padded)
    const int row = t - 32;
    float s = 0.f;
    for (int j = 0; j < 128; ++j) s += (float)Bv[row][j];
    vs_ws[row] = s;
  }
}

// ---------------- Kernel 2: one workgroup (4 wave32) per cell ----------------
__global__ __launch_bounds__(128) void pillar_cell(
    const float* __restrict__ pts_g, const int* __restrict__ mask_g,
    const float* __restrict__ pos,
    const float* __restrict__ qw, const float* __restrict__ qb,
    const float* __restrict__ ww, const float* __restrict__ wb,
    const float* __restrict__ gk_ws, const float* __restrict__ vs_ws,
    float* __restrict__ out) {
  __shared__ _Float16 Aq[32][128];  // augmented q-weights, rows cm, cols i
  __shared__ _Float16 Bq[16][128];  // Bq[n][i] = mask[i]*ww[n*128+i] (n<4)
  __shared__ float QT[32][16];
  __shared__ float GKs[20][20];
  __shared__ float VSs[20];
  __shared__ float Wneg[4];
  __shared__ float wts[P_];
  __shared__ int   selb[P_];
  __shared__ int   mk[P_];
  __shared__ float4 ptsL[P_];
  __shared__ float4 opts[KSEL_];
  __shared__ float  ow[KSEL_];

  const int cell = blockIdx.x;     // flattened (b,h,w)
  const int t = threadIdx.x;       // 0..127 == point index p
  const int lane = t & 31, wid = t >> 5;

  // per-thread point, mask, augmented coordinates
  const float4 pt = reinterpret_cast<const float4*>(pts_g + (size_t)cell * P_ * C_)[t];
  const int mp = (mask_g[cell * P_ + t] != 0);
  float x[5];
  x[0] = pt.x + pos[t * 4 + 0];
  x[1] = pt.y + pos[t * 4 + 1];
  x[2] = pt.z + pos[t * 4 + 2];
  x[3] = pt.w + pos[t * 4 + 3];
  x[4] = 1.0f;
  ptsL[t] = pt;
  mk[t] = mp;

  {  // stage Aq
    const int row = t & 31, jb = (t >> 5) * 32;
    const int c = row >> 2, m = row & 3;
    for (int j = jb; j < jb + 32; ++j) {
      float v = 0.f;
      if (row < 20) v = (c < 4) ? qw[c * HID_ + m * HD_ + j] : qb[m * HD_ + j];
      Aq[row][j] = (_Float16)v;
    }
  }
  // stage Bq (mask folded into the B operand)
#pragma unroll
  for (int n = 0; n < 16; ++n)
    Bq[n][t] = (_Float16)((n < 4 && mp) ? ww[n * HD_ + t] : 0.f);

  // bring cell-invariant GK / VS into LDS
  for (int idx = t; idx < 400; idx += 128)
    GKs[idx / 20][idx % 20] = gk_ws[(idx / 20) * 32 + (idx % 20)];
  if (t < 20) VSs[t] = vs_ws[t];
  if (t < 4)  Wneg[t] = 0.f;
  __syncthreads();

  // Wneg[n] = sum over masked-off i of ww[n*128+i]  (LDS ds_add_f32 atomics)
  if (!mp) {
#pragma unroll
    for (int n = 0; n < 4; ++n) atomicAdd(&Wneg[n], ww[n * HD_ + t]);
  }

  // QT[20x4] via WMMA: waves 0/1 each own one M-tile, 4 k-steps
  if (wid < 2) {
    const int mt = wid;
    v8f acc = {};
#pragma unroll
    for (int kc = 0; kc < 4; ++kc) {
      v16h a = ldsA_frag(&Aq[0][0], 128, mt * 16, kc * 32, lane);
      v16h b = ldsBT_frag(&Bq[0][0], 128, 0, kc * 32, lane);
      acc = __builtin_amdgcn_wmma_f32_16x16x32_f16(false, a, false, b,
                                                   (short)0, acc, false, false);
    }
    const int h = lane >> 4, n = lane & 15;
#pragma unroll
    for (int r = 0; r < 8; ++r) QT[mt * 16 + r + 8 * h][n] = acc[r];
  }
  __syncthreads();

  // per-point scalar epilogue (~500 MACs/thread)
  float sMN = 0.f;
#pragma unroll
  for (int m = 0; m < 4; ++m) {
#pragma unroll
    for (int n = 0; n < 4; ++n) {
      float T = 0.f, G = 0.f;
#pragma unroll
      for (int c = 0; c < 5; ++c) {
        T += x[c] * QT[c * 4 + m][n];
        float g = 0.f;
#pragma unroll
        for (int c2 = 0; c2 < 5; ++c2) g += x[c2] * GKs[c * 4 + m][c2 * 4 + n];
        G += x[c] * g;
      }
      sMN += T * G;
    }
  }
  float negterm = 0.f;
#pragma unroll
  for (int n = 0; n < 4; ++n) {
    float S = 0.f;
#pragma unroll
    for (int c = 0; c < 5; ++c) S += x[c] * VSs[c * 4 + n];
    negterm += S * Wneg[n];
  }
  const float inv_s = 0.088388347648318447f;  // 1/sqrt(128)
  const float ad = inv_s * sMN + NEG_ * negterm + wb[0];
  const float wgt = (1.f / (1.f + __expf(-ad))) * (mp ? 1.f : 0.f);
  wts[t] = wgt;
  selb[t] = (wgt > THRESH_) && mp;
  __syncthreads();

  // exact selection: stable ranks matching lax.top_k / argsort semantics
  int cnt = 0, rank_asc = 0, rank_top = 0;
  const float wp = wts[t];
  for (int q = 0; q < P_; ++q) {
    const int sq = selb[q];
    cnt += sq;
    if (sq && q < t) ++rank_asc;
    if (sq && (wts[q] > wp || (wts[q] == wp && q < t))) ++rank_top;
  }
  if (t < KSEL_) { opts[t] = make_float4(0.f, 0.f, 0.f, 0.f); ow[t] = 0.f; }
  __syncthreads();

  if (selb[t]) {
    const int slot = (cnt > KSEL_) ? rank_top : rank_asc;
    if (slot < KSEL_) { opts[slot] = ptsL[t]; ow[slot] = wts[t]; }
  }
  if (t == 0 && cnt == 0) {  // fallback: first valid point, slot 0 only
    int fv = -1;
    for (int q = 0; q < P_; ++q) if (mk[q]) { fv = q; break; }
    if (fv >= 0) { opts[0] = ptsL[fv]; ow[0] = wts[fv]; }
  }
  __syncthreads();

  if (t < KSEL_) {
    reinterpret_cast<float4*>(out + (size_t)cell * KSEL_ * C_)[t] = opts[t];
    out[NCELL_ * KSEL_ * C_ + cell * KSEL_ + t] = ow[t];
  }
}

extern "C" void kernel_launch(void* const* d_in, const int* in_sizes, int n_in,
                              void* d_out, int out_size, void* d_ws, size_t ws_size,
                              hipStream_t stream) {
  (void)in_sizes; (void)n_in; (void)out_size; (void)ws_size;
  const float* pts = (const float*)d_in[0];
  const int*   msk = (const int*)d_in[1];   // pillar_masks (bool -> int per harness)
  const float* pos = (const float*)d_in[2];
  const float* qw  = (const float*)d_in[3];
  const float* qb  = (const float*)d_in[4];
  const float* kw  = (const float*)d_in[5];
  const float* kb  = (const float*)d_in[6];
  const float* vw  = (const float*)d_in[7];
  const float* vb  = (const float*)d_in[8];
  const float* ww  = (const float*)d_in[9];
  const float* wb  = (const float*)d_in[10];

  float* gk_ws = (float*)d_ws;        // 32*32 floats
  float* vs_ws = gk_ws + 32 * 32;     // 32 floats

  pillar_precompute<<<1, 64, 0, stream>>>(kw, kb, vw, vb, gk_ws, vs_ws);
  pillar_cell<<<NCELL_, 128, 0, stream>>>(pts, msk, pos, qw, qb, ww, wb,
                                          gk_ws, vs_ws, (float*)d_out);
}
